// GraphspaAttentionLayer_6476810683007
// MI455X (gfx1250) — matrix operand order repaired and checked by reading, not compiled, not run on hardware
//
#include <hip/hip_runtime.h>
#include <hip/hip_bf16.h>
#include <cstdint>

#define IN_F  256
#define OUT_F 256
#define NB    8
#define NN    2048
#define ALPHA 0.2f

typedef __attribute__((ext_vector_type(16))) __bf16 v16bf;
typedef __attribute__((ext_vector_type(8)))  __bf16 v8bf;
typedef __attribute__((ext_vector_type(8)))  float  v8f;

__device__ __forceinline__ float leaky(float x) { return x > 0.f ? x : ALPHA * x; }

// ---------------- u1 = W1 @ a[:256], u2 = W1 @ a[256:] ----------------
__global__ void k_prep_u(const float* __restrict__ W1, const float* __restrict__ a,
                         float* __restrict__ u1, float* __restrict__ u2) {
  int c = threadIdx.x;
  const float* row = W1 + c * IN_F;
  float s1 = 0.f, s2 = 0.f;
  for (int o = 0; o < IN_F; ++o) {
    float w = row[o];
    s1 += w * a[o];
    s2 += w * a[IN_F + o];
  }
  u1[c] = s1; u2[c] = s2;
}

// ---------------- bf16 transposed weights: Wtb[o][c] = W[c][o] ----------------
__global__ void k_transpose_w(const float* __restrict__ W1, const float* __restrict__ W2,
                              __bf16* __restrict__ W1tb, __bf16* __restrict__ W2tb) {
  for (int e = blockIdx.x * blockDim.x + threadIdx.x; e < 2 * IN_F * IN_F;
       e += gridDim.x * blockDim.x) {
    int m = e >> 16, rem = e & 65535;
    int c = rem >> 8, o = rem & 255;
    if (m == 0) W1tb[o * IN_F + c] = (__bf16)W1[c * IN_F + o];
    else        W2tb[o * IN_F + c] = (__bf16)W2[c * IN_F + o];
  }
}

// ---------------- Wh1[g]=dot(h[g],u1), Wh2[g]=dot(h[g],u2); one wave per row ----------------
__global__ void k_rowdots(const float* __restrict__ h, const float* __restrict__ u1,
                          const float* __restrict__ u2, float* __restrict__ Wh1,
                          float* __restrict__ Wh2) {
  int wave = threadIdx.x >> 5, lane = threadIdx.x & 31;
  long g = (long)blockIdx.x * 8 + wave;                 // 16384 rows
  const float* hp = h + g * IN_F + lane * 8;
  float4 x0 = *(const float4*)hp;
  float4 x1 = *(const float4*)(hp + 4);
  const float* u1p = u1 + lane * 8;
  const float* u2p = u2 + lane * 8;
  float4 a0 = *(const float4*)u1p, a1 = *(const float4*)(u1p + 4);
  float4 b0 = *(const float4*)u2p, b1 = *(const float4*)(u2p + 4);
  float s1 = x0.x*a0.x + x0.y*a0.y + x0.z*a0.z + x0.w*a0.w
           + x1.x*a1.x + x1.y*a1.y + x1.z*a1.z + x1.w*a1.w;
  float s2 = x0.x*b0.x + x0.y*b0.y + x0.z*b0.z + x0.w*b0.w
           + x1.x*b1.x + x1.y*b1.y + x1.z*b1.z + x1.w*b1.w;
  for (int off = 16; off >= 1; off >>= 1) {
    s1 += __shfl_xor(s1, off, 32);
    s2 += __shfl_xor(s2, off, 32);
  }
  if (lane == 0) { Wh1[g] = s1; Wh2[g] = s2; }
}

// ---------------- per-batch max of Wh2 -> S[b] ----------------
__global__ void k_batchmax(const float* __restrict__ Wh2, float* __restrict__ S) {
  __shared__ float red[256];
  int b = blockIdx.x, t = threadIdx.x;
  float m = -1e30f;
  for (int k = t; k < NN; k += 256) m = fmaxf(m, Wh2[b * NN + k]);
  red[t] = m; __syncthreads();
  for (int s = 128; s > 0; s >>= 1) {
    if (t < s) red[t] = fmaxf(red[t], red[t + s]);
    __syncthreads();
  }
  if (t == 0) S[b] = red[0];
}

// ---------------- projection GEMM: dst^T[(b,f),node] = bf16( src @ W ) ----------------
// One wave computes a 16-row x 256-col stripe (16 WMMA tiles, K=256 in 8 steps of 32).
__global__ void __launch_bounds__(256) k_project(
    const float* __restrict__ h, const float* __restrict__ l1,
    const __bf16* __restrict__ W1tb, const __bf16* __restrict__ W2tb,
    __bf16* __restrict__ WhT, __bf16* __restrict__ WlT) {
  const int sel = blockIdx.y;
  const float*  src = sel ? l1 : h;
  const __bf16* Wtb = sel ? W2tb : W1tb;
  __bf16*       dst = sel ? WlT : WhT;

  int wave = threadIdx.x >> 5, lane = threadIdx.x & 31;
  int r = lane & 15, hi = lane >> 4;
  long i0 = (long)blockIdx.x * 128 + wave * 16;

  v8f acc[16];
  v8f zero = {0.f, 0.f, 0.f, 0.f, 0.f, 0.f, 0.f, 0.f};
#pragma unroll
  for (int t = 0; t < 16; ++t) acc[t] = zero;

  const float* hp = src + (i0 + r) * IN_F;
  const __bf16* wp = Wtb + r * IN_F;     // + t*16*IN_F + ks*32 + hi*16
#pragma unroll
  for (int ks = 0; ks < 8; ++ks) {
    int k0 = ks * 32 + hi * 16;
    float4 f0 = *(const float4*)(hp + k0);
    float4 f1 = *(const float4*)(hp + k0 + 4);
    float4 f2 = *(const float4*)(hp + k0 + 8);
    float4 f3 = *(const float4*)(hp + k0 + 12);
    v16bf a;
    a[0] = (__bf16)f0.x; a[1] = (__bf16)f0.y; a[2]  = (__bf16)f0.z; a[3]  = (__bf16)f0.w;
    a[4] = (__bf16)f1.x; a[5] = (__bf16)f1.y; a[6]  = (__bf16)f1.z; a[7]  = (__bf16)f1.w;
    a[8] = (__bf16)f2.x; a[9] = (__bf16)f2.y; a[10] = (__bf16)f2.z; a[11] = (__bf16)f2.w;
    a[12]= (__bf16)f3.x; a[13]= (__bf16)f3.y; a[14] = (__bf16)f3.z; a[15] = (__bf16)f3.w;

    v16bf Bb[2];
    Bb[0] = *(const v16bf*)(wp + ks * 32 + hi * 16);   // tile t=0
#pragma unroll
    for (int t = 0; t < 16; ++t) {
      if (t < 15)
        Bb[(t + 1) & 1] = *(const v16bf*)(wp + (t + 1) * 16 * IN_F + ks * 32 + hi * 16);
      acc[t] = __builtin_amdgcn_wmma_f32_16x16x32_bf16(
          false, a, false, Bb[t & 1], (short)0, acc[t], false, false);
    }
  }

  int b = (int)(i0 / NN);
  int node0 = (int)(i0 - (long)b * NN) + 8 * hi;
#pragma unroll
  for (int t = 0; t < 16; ++t) {
    int f = t * 16 + r;
    __bf16* dp = dst + ((long)b * IN_F + f) * NN + node0;
    v8bf pack;
#pragma unroll
    for (int v = 0; v < 8; ++v) pack[v] = (__bf16)acc[t][v];
    *(v8bf*)dp = pack;   // 8 consecutive nodes, 16B store
  }
}

// ---------------- fused attention: scores -> masked softmax (global bound) -> P@V, ELU ----------------
// Block = 16 query rows of one batch; 8 waves: sel = wave/4 picks output, (wave&3)*64 picks feat slice.
// Software-pipelined: adj/Wh2/V-fragment loads double-buffered one key-block ahead.
#define PCOMP(idx, AV, WV)                                           \
  { float e_ = wh1r + (WV); e_ = e_ > 0.f ? e_ : ALPHA * e_;         \
    float p_ = ((AV) > 0) ? __expf(e_ - mrow) : 0.f;                 \
    lsum += p_; afrag[idx] = (__bf16)p_; }

__global__ void __launch_bounds__(256) k_attn(
    const int* __restrict__ adj, const float* __restrict__ Wh1,
    const float* __restrict__ Wh2, const float* __restrict__ S,
    const __bf16* __restrict__ WhT, const __bf16* __restrict__ WlT,
    float* __restrict__ out) {
  int b  = blockIdx.x >> 7;
  int i0 = (blockIdx.x & 127) << 4;
  int wave = threadIdx.x >> 5, lane = threadIdx.x & 31;
  int r = lane & 15, hi = lane >> 4;
  int sel = wave >> 2;
  int f0  = (wave & 3) * 64;

  const __bf16* V = (sel ? WlT : WhT) + (long)b * IN_F * NN;
  float wh1r = Wh1[b * NN + i0 + r];
  float mrow = leaky(wh1r + S[b]);
  const int*   adjp = adj + (long)(b * NN + i0 + r) * NN;
  const float* wh2p = Wh2 + b * NN;

  v8f acc[4];
  v8f zero = {0.f, 0.f, 0.f, 0.f, 0.f, 0.f, 0.f, 0.f};
#pragma unroll
  for (int t = 0; t < 4; ++t) acc[t] = zero;
  float lsum = 0.f;

  // ping-pong staging registers
  int4   Ab[2][4];
  float4 Wb[2][4];
  v16bf  Bb[2][4];

  auto load_stage = [&](int s, int j0_) {
    int ka = j0_ + hi * 16;
    Ab[s][0] = *(const int4*)(adjp + ka);
    Ab[s][1] = *(const int4*)(adjp + ka + 4);
    Ab[s][2] = *(const int4*)(adjp + ka + 8);
    Ab[s][3] = *(const int4*)(adjp + ka + 12);
    Wb[s][0] = *(const float4*)(wh2p + ka);
    Wb[s][1] = *(const float4*)(wh2p + ka + 4);
    Wb[s][2] = *(const float4*)(wh2p + ka + 8);
    Wb[s][3] = *(const float4*)(wh2p + ka + 12);
#pragma unroll
    for (int t = 0; t < 4; ++t)
      Bb[s][t] = *(const v16bf*)(V + (long)(f0 + t * 16 + r) * NN + j0_ + hi * 16);
    __builtin_prefetch((const void*)(adjp + ka + 32), 0, 1);
  };

  auto compute_stage = [&](int s) {
    v16bf afrag;
    PCOMP(0,  Ab[s][0].x, Wb[s][0].x) PCOMP(1,  Ab[s][0].y, Wb[s][0].y)
    PCOMP(2,  Ab[s][0].z, Wb[s][0].z) PCOMP(3,  Ab[s][0].w, Wb[s][0].w)
    PCOMP(4,  Ab[s][1].x, Wb[s][1].x) PCOMP(5,  Ab[s][1].y, Wb[s][1].y)
    PCOMP(6,  Ab[s][1].z, Wb[s][1].z) PCOMP(7,  Ab[s][1].w, Wb[s][1].w)
    PCOMP(8,  Ab[s][2].x, Wb[s][2].x) PCOMP(9,  Ab[s][2].y, Wb[s][2].y)
    PCOMP(10, Ab[s][2].z, Wb[s][2].z) PCOMP(11, Ab[s][2].w, Wb[s][2].w)
    PCOMP(12, Ab[s][3].x, Wb[s][3].x) PCOMP(13, Ab[s][3].y, Wb[s][3].y)
    PCOMP(14, Ab[s][3].z, Wb[s][3].z) PCOMP(15, Ab[s][3].w, Wb[s][3].w)
#pragma unroll
    for (int t = 0; t < 4; ++t)
      acc[t] = __builtin_amdgcn_wmma_f32_16x16x32_bf16(
          false, afrag, false, Bb[s][t], (short)0, acc[t], false, false);
  };

  load_stage(0, 0);
  for (int it = 0; it < 64; it += 2) {
    int j0 = it * 32;
    load_stage(1, j0 + 32);       // stage next while computing current
    compute_stage(0);
    if (it + 2 < 64) load_stage(0, j0 + 64);
    compute_stage(1);
  }

  lsum += __shfl_xor(lsum, 16, 32);   // combine the two half-row partials
  float inv[8];
#pragma unroll
  for (int v = 0; v < 8; ++v) inv[v] = 1.f / __shfl(lsum, v + 8 * hi, 32);

  long outbase = (long)sel * NB * NN * OUT_F + (long)b * NN * OUT_F;
#pragma unroll
  for (int t = 0; t < 4; ++t) {
    int f = f0 + t * 16 + r;
#pragma unroll
    for (int v = 0; v < 8; ++v) {
      int row = v + 8 * hi;
      float x = acc[t][v] * inv[v];
      x = x > 0.f ? x : (__expf(x) - 1.f);   // ELU (alpha=1)
      out[outbase + (long)(i0 + row) * OUT_F + f] = x;
    }
  }
}

extern "C" void kernel_launch(void* const* d_in, const int* in_sizes, int n_in,
                              void* d_out, int out_size, void* d_ws, size_t ws_size,
                              hipStream_t stream) {
  const float* h   = (const float*)d_in[0];
  const int*   adj = (const int*)d_in[1];
  const float* l1  = (const float*)d_in[2];
  const float* W1  = (const float*)d_in[3];
  const float* W2  = (const float*)d_in[4];
  const float* a   = (const float*)d_in[5];
  float* out = (float*)d_out;
  char*  ws  = (char*)d_ws;

  float*  u1   = (float*)(ws + 0);
  float*  u2   = (float*)(ws + 4096);
  float*  S    = (float*)(ws + 8192);
  float*  Wh1  = (float*)(ws + 16384);
  float*  Wh2  = (float*)(ws + 16384 + 65536);
  __bf16* W1tb = (__bf16*)(ws + 147456);
  __bf16* W2tb = (__bf16*)(ws + 147456 + 131072);
  __bf16* WhT  = (__bf16*)(ws + 409600);
  __bf16* WlT  = (__bf16*)(ws + 409600 + 8388608);

  k_prep_u     <<<1,    256, 0, stream>>>(W1, a, u1, u2);
  k_transpose_w<<<128,  256, 0, stream>>>(W1, W2, W1tb, W2tb);
  k_rowdots    <<<2048, 256, 0, stream>>>(h, u1, u2, Wh1, Wh2);
  k_batchmax   <<<8,    256, 0, stream>>>(Wh2, S);
  k_project    <<<dim3(128, 2), 256, 0, stream>>>(h, l1, W1tb, W2tb, WhT, WlT);
  k_attn       <<<1024, 256, 0, stream>>>(adj, Wh1, Wh2, S, WhT, WlT, out);
}
